// EMolES_70866960384261
// MI455X (gfx1250) — matrix operand order repaired and checked by reading, not compiled, not run on hardware
//
#include <hip/hip_runtime.h>

#define NN 16384
#define NE 262144
#define EPS 1e-8f

typedef __attribute__((ext_vector_type(16))) _Float16 v16h;
typedef __attribute__((ext_vector_type(8)))  _Float16 v8h;
typedef __attribute__((ext_vector_type(8)))  float    v8f;
typedef __attribute__((ext_vector_type(4)))  unsigned int u32x4;
typedef __attribute__((ext_vector_type(8)))  int      i32x8;
typedef __attribute__((ext_vector_type(4)))  int      i32x4;

#if defined(__has_builtin)
# if __has_builtin(__builtin_amdgcn_tensor_load_to_lds) && __has_builtin(__builtin_amdgcn_s_wait_tensorcnt)
#  define USE_TDM 1
# endif
#endif
#ifndef USE_TDM
# define USE_TDM 0
#endif

union HF16 { v16h v; v8h h[2]; };

// ---- WMMA fragment helpers (wave32, 16x16x32 f16 -> f32) -------------------
__device__ __forceinline__ v16h load_a_frag(const _Float16* base, int strideH, int lane, int ktile) {
  int m  = lane & 15;
  int kb = ktile * 32 + ((lane & 16) ? 8 : 0);
  HF16 f;
  f.h[0] = *(const v8h*)(base + m * strideH + kb);
  f.h[1] = *(const v8h*)(base + m * strideH + kb + 16);
  return f.v;
}
__device__ __forceinline__ v16h load_b_frag(const _Float16* wfrag, int tile, int lane) {
  const _Float16* p = wfrag + (size_t)tile * 512 + lane * 16;
  HF16 f;
  f.h[0] = *(const v8h*)(p);
  f.h[1] = *(const v8h*)(p + 8);
  return f.v;
}
__device__ __forceinline__ void store_acc(float* base, int strideF, int lane, int nt, v8f acc, float scale) {
  int n  = nt * 16 + (lane & 15);
  int mb = (lane & 16) ? 8 : 0;
#pragma unroll
  for (int j = 0; j < 8; ++j) base[(mb + j) * strideF + n] = acc[j] * scale;
}

#define WMMA(A,B,C) __builtin_amdgcn_wmma_f32_16x16x32_f16(false,(A),false,(B),(short)0,(C),false,false)
#define ZERO8 {0.f,0.f,0.f,0.f,0.f,0.f,0.f,0.f}

__device__ __forceinline__ float sigm(float x){ return 1.f/(1.f+__expf(-x)); }

#if USE_TDM
// 1-D contiguous DMA: global -> LDS via Tensor Data Mover (D# per ISA 08 §8)
// clang-23 toolchain: 6-arg builtin (u32x4, i32x8, i32x4, i32x4, i32x8, i32 cpol)
__device__ __forceinline__ void tdm_load_1d(unsigned lds_addr, const void* gptr, unsigned n4B) {
  unsigned long long ga = (unsigned long long)(size_t)gptr;
  u32x4 g0;
  g0[0] = 1u;                                            // count=1, user descriptor
  g0[1] = lds_addr;                                      // LDS byte address
  g0[2] = (unsigned)(ga & 0xffffffffull);                // global_addr lo
  g0[3] = (unsigned)((ga >> 32) & 0x01ffffffull) | 0x80000000u; // global_addr hi | type=2
  i32x8 g1;
  g1[0] = 0x00020000;                                    // data_size=4B; no mask/flags
  g1[1] = (int)((n4B & 0xffffu) << 16);                  // tensor_dim0[15:0]
  g1[2] = (int)(((n4B >> 16) & 0xffffu) | (1u << 16));   // tensor_dim0[31:16] | tensor_dim1=1
  g1[3] = (int)((n4B & 0xffffu) << 16);                  // tile_dim0
  g1[4] = 1;                                             // tile_dim1=1, tile_dim2=0
  g1[5] = (int)n4B;                                      // tensor_dim0_stride lo
  g1[6] = 0;
  g1[7] = 0;
  i32x4 z4 = {0, 0, 0, 0};
  i32x8 z8 = {0, 0, 0, 0, 0, 0, 0, 0};
  __builtin_amdgcn_tensor_load_to_lds(g0, g1, z4, z4, z8, 0);
}
#endif

// ---- weight conversion: row-major fp32 (K x Ncols) -> f16 B-fragment order --
__global__ void convert_frag_kernel(const float* __restrict__ W, _Float16* __restrict__ dst,
                                    int KT, int NT, int Ncols, int negate) {
  int total = KT * NT * 512;
  for (int idx = blockIdx.x * blockDim.x + threadIdx.x; idx < total; idx += gridDim.x * blockDim.x) {
    int h    = idx & 15;
    int lane = (idx >> 4) & 31;
    int tile = idx >> 9;
    int nt   = tile % NT;
    int kt   = tile / NT;
    int K    = kt * 32 + ((lane & 16) ? 8 : 0) + (h < 8 ? h : h + 8);
    int col  = nt * 16 + (lane & 15);
    float v  = W[(size_t)K * Ncols + col];
    dst[idx] = (_Float16)(negate ? -v : v);
  }
}

// ---- node separable layernorm (1 wave per node) ----------------------------
__global__ __launch_bounds__(256) void node_ln_kernel(const float* __restrict__ nf,
    const float* __restrict__ g0, const float* __restrict__ b0, const float* __restrict__ g1,
    float* __restrict__ nodeS, float* __restrict__ nodeV) {
  int lane = threadIdx.x & 31;
  int w    = threadIdx.x >> 5;
  int n    = blockIdx.x * 8 + w;
  const float* row = nf + (size_t)n * 320;
  float v[4]; float sum = 0.f, ss = 0.f;
#pragma unroll
  for (int j = 0; j < 4; ++j) { float x = row[lane*4+j]; v[j]=x; sum+=x; ss+=x*x; }
  for (int o = 16; o; o >>= 1) { sum += __shfl_xor(sum,o); ss += __shfl_xor(ss,o); }
  float mu  = sum * (1.f/128.f);
  float var = ss  * (1.f/128.f) - mu*mu;
  float rs  = rsqrtf(var + EPS);
#pragma unroll
  for (int j = 0; j < 4; ++j) { int c = lane*4+j; nodeS[(size_t)n*128+c] = (v[j]-mu)*rs*g0[c]+b0[c]; }
  float vv[6]; float s2 = 0.f;
#pragma unroll
  for (int j = 0; j < 6; ++j) { float x = row[128 + lane*6+j]; vv[j]=x; s2+=x*x; }
  for (int o = 16; o; o >>= 1) s2 += __shfl_xor(s2,o);
  float rsv = rsqrtf(s2 * (1.f/192.f) + EPS);
#pragma unroll
  for (int j = 0; j < 6; ++j) { int q = lane*6+j; nodeV[(size_t)n*192+q] = vv[j]*rsv*g1[q/3]; }
}

// ---- fused per-edge kernel: 32 edges per 256-thread block ------------------
__global__ __launch_bounds__(256) void edge_kernel(
    const float* __restrict__ edge_feat, const float* __restrict__ edge_vec,
    const float* __restrict__ latents,
    const float* __restrict__ g0e, const float* __restrict__ b0e, const float* __restrict__ g1e,
    const int*   __restrict__ edge_index,
    const float* __restrict__ nodeS, const float* __restrict__ nodeV,
    const _Float16* __restrict__ Wm0h, const _Float16* __restrict__ wrh,
    const _Float16* __restrict__ wih,  const _Float16* __restrict__ wihn,
    const _Float16* __restrict__ Wp0h, const _Float16* __restrict__ Wenvh,
    const _Float16* __restrict__ Wpvh, const float* __restrict__ bp0,
    float* __restrict__ sa, float* __restrict__ va) {

  // ovl: phase1/2 Xh(32x488 h) + c1h/c2h(32x168 h)  <->  phase4/5 s2L(32x128 f) + wL(32x192 f)
  __shared__ __align__(16) char ovl[52736];
  __shared__ __align__(16) float    frameL[32*12];
  __shared__ int srcL[32];
  __shared__ int dstL[32];
  __shared__ __align__(16) float    sOutL[32*256];   // phase0/1: overlaid by efS (32x160 f)
  __shared__ __align__(16) float    c12oL[32*128];   // phase0/1: overlaid by latF (32x128 f)
  __shared__ __align__(16) _Float16 sActH[32*136];
  __shared__ __align__(16) _Float16 latH [32*136];
  __shared__ __align__(16) _Float16 vgH  [96*72];
  __shared__ __align__(16) float    v2L  [96*64];

  _Float16* Xh  = (_Float16*)ovl;                         // 32*488 halves
  _Float16* c1h = (_Float16*)(ovl + 32*488*2);            // 32*168 halves
  _Float16* c2h = (_Float16*)(ovl + 32*488*2 + 32*168*2);
  float*    s2L = (float*)ovl;                            // 32*128 f32
  float*    wL  = (float*)(ovl + 32*128*4);               // 32*192 f32
  float*    efS = sOutL;                                  // 32*160 f32 staging
  float*    latF = c12oL;                                 // 32*128 f32 staging

  const int tid  = threadIdx.x;
  const int lane = tid & 31;
  const int w    = tid >> 5;
  const int e0   = blockIdx.x * 32;

  // ---- Phase 0: stage edge_feat + latents slabs into LDS; frames + indices ----
#if USE_TDM
  if (tid < 32) {
    tdm_load_1d((unsigned)(size_t)efS,  edge_feat + (size_t)e0 * 160, 32 * 160);
    tdm_load_1d((unsigned)(size_t)latF, latents   + (size_t)e0 * 128, 32 * 128);
  }
#else
  for (int idx = tid; idx < 32 * 160; idx += 256) efS[idx]  = edge_feat[(size_t)e0 * 160 + idx];
  for (int idx = tid; idx < 32 * 128; idx += 256) latF[idx] = latents[(size_t)e0 * 128 + idx];
#endif
  if (tid < 32) {
    int e = e0 + tid;
    float vx = edge_vec[e*3+0], vy = edge_vec[e*3+1], vz = edge_vec[e*3+2];
    float rn = rsqrtf(vx*vx + vy*vy + vz*vz + EPS);
    float ax = vx*rn, ay = vy*rn, az = vz*rn;
    float hx, hy, hz;
    if (fabsf(az) < 0.99f) { hx=0.f; hy=0.f; hz=1.f; } else { hx=1.f; hy=0.f; hz=0.f; }
    float e1x = hy*az - hz*ay, e1y = hz*ax - hx*az, e1z = hx*ay - hy*ax;
    float rn1 = rsqrtf(e1x*e1x + e1y*e1y + e1z*e1z + EPS);
    e1x*=rn1; e1y*=rn1; e1z*=rn1;
    float e2x = ay*e1z - az*e1y, e2y = az*e1x - ax*e1z, e2z = ax*e1y - ay*e1x;
    float* fr = frameL + tid*12;
    fr[0]=ax; fr[1]=ay; fr[2]=az; fr[3]=e1x; fr[4]=e1y; fr[5]=e1z; fr[6]=e2x; fr[7]=e2y; fr[8]=e2z;
    srcL[tid] = edge_index[e];
    dstL[tid] = edge_index[NE + e];
  }
#if USE_TDM
  if (tid < 32) __builtin_amdgcn_s_wait_tensorcnt(0);
#endif
  __syncthreads();

  // ---- Phase 1: edge LN + gathers + frame projections -> LDS f16 ----
  {
    const int sub = tid & 7;         // 8 threads per edge
    const int le  = tid >> 3;        // local edge 0..31
    const int src = srcL[le], dst = dstL[le];
    const float* ef = efS + le * 160;
    // edge scalar LN (64) -> Xh cols 128..191
    {
      float vals[8]; float sum = 0.f, ss = 0.f;
#pragma unroll
      for (int j = 0; j < 8; ++j) { float x = ef[sub*8 + j]; vals[j]=x; sum+=x; ss+=x*x; }
      for (int o = 4; o; o >>= 1) { sum += __shfl_xor(sum,o); ss += __shfl_xor(ss,o); }
      float mu  = sum * (1.f/64.f);
      float var = ss  * (1.f/64.f) - mu*mu;
      float rs  = rsqrtf(var + EPS);
#pragma unroll
      for (int j = 0; j < 8; ++j) { int c = sub*8+j; Xh[le*488 + 128 + c] = (_Float16)((vals[j]-mu)*rs*g0e[c] + b0e[c]); }
    }
    // edge vector rms (32 x 3)
    float rsv;
    {
      float ss = 0.f;
#pragma unroll
      for (int j = 0; j < 12; ++j) { float x = ef[64 + sub*12 + j]; ss += x*x; }
      for (int o = 4; o; o >>= 1) ss += __shfl_xor(ss,o);
      rsv = rsqrtf(ss * (1.f/96.f) + EPS);
    }
    // gathered (pre-normalized) node scalars -> Xh cols 0..127 / 192..319
#pragma unroll
    for (int j = 0; j < 16; ++j) {
      Xh[le*488 +       sub*16 + j] = (_Float16)nodeS[(size_t)src*128 + sub*16 + j];
      Xh[le*488 + 192 + sub*16 + j] = (_Float16)nodeS[(size_t)dst*128 + sub*16 + j];
    }
    // v_cat projections -> c0 (Xh 320..479), c1h, c2h
    const float* fr = frameL + le*12;
    float ax=fr[0],ay=fr[1],az=fr[2],e1x=fr[3],e1y=fr[4],e1z=fr[5],e2x=fr[6],e2y=fr[7],e2z=fr[8];
#pragma unroll
    for (int k = 0; k < 20; ++k) {
      int ch = sub + 8*k;            // 0..159
      float vx, vy, vz;
      if (ch < 64)       { const float* p = nodeV + (size_t)src*192 + ch*3; vx=p[0]; vy=p[1]; vz=p[2]; }
      else if (ch < 96)  { int u = ch-64; const float* p = ef + 64 + u*3;
                           float s = rsv * g1e[u]; vx=p[0]*s; vy=p[1]*s; vz=p[2]*s; }
      else               { const float* p = nodeV + (size_t)dst*192 + (ch-96)*3; vx=p[0]; vy=p[1]; vz=p[2]; }
      Xh [le*488 + 320 + ch] = (_Float16)(vx*ax  + vy*ay  + vz*az);
      c1h[le*168 + ch]       = (_Float16)(vx*e1x + vy*e1y + vz*e1z);
      c2h[le*168 + ch]       = (_Float16)(vx*e2x + vy*e2y + vz*e2z);
    }
  }
  // latents f32 (LDS) -> f16 LDS
  for (int idx = tid; idx < 32*128; idx += 256) {
    int e = idx >> 7, k = idx & 127;
    latH[e*136 + k] = (_Float16)latF[idx];
  }
  __syncthreads();

  // ---- Phase 2: WMMA  m0 = X @ W_m0 (K=480,N=256), 2 M-tiles share B frags ----
  {
    v8f a00 = ZERO8, a01 = ZERO8, a10 = ZERO8, a11 = ZERO8;
    int nt0 = 2*w, nt1 = 2*w + 1;
#pragma unroll
    for (int kt = 0; kt < 15; ++kt) {
      v16h af0 = load_a_frag(Xh,          488, lane, kt);
      v16h af1 = load_a_frag(Xh + 16*488, 488, lane, kt);
      v16h b0  = load_b_frag(Wm0h, kt*16 + nt0, lane);
      v16h b1  = load_b_frag(Wm0h, kt*16 + nt1, lane);
      a00 = WMMA(af0, b0, a00);  a01 = WMMA(af0, b1, a01);
      a10 = WMMA(af1, b0, a10);  a11 = WMMA(af1, b1, a11);
    }
    const float m0s = 0.04564354646f;   // 1/sqrt(480)
    store_acc(sOutL,          256, lane, nt0, a00, m0s);
    store_acc(sOutL,          256, lane, nt1, a01, m0s);
    store_acc(sOutL + 16*256, 256, lane, nt0, a10, m0s);
    store_acc(sOutL + 16*256, 256, lane, nt1, a11, m0s);
  }
  // rotation: waves 0..3 -> c1o = c1@w_r + c2@(-w_i); waves 4..7 -> c2o = c1@w_i + c2@w_r
  {
    v8f acc0 = ZERO8, acc1 = ZERO8;
    int nt = w & 3;
    const _Float16* B1 = (w < 4) ? wrh  : wih;
    const _Float16* B2 = (w < 4) ? wihn : wrh;
#pragma unroll
    for (int kt = 0; kt < 5; ++kt) {
      v16h b1 = load_b_frag(B1, kt*4 + nt, lane);
      v16h b2 = load_b_frag(B2, kt*4 + nt, lane);
      acc0 = WMMA(load_a_frag(c1h,          168, lane, kt), b1, acc0);
      acc0 = WMMA(load_a_frag(c2h,          168, lane, kt), b2, acc0);
      acc1 = WMMA(load_a_frag(c1h + 16*168, 168, lane, kt), b1, acc1);
      acc1 = WMMA(load_a_frag(c2h + 16*168, 168, lane, kt), b2, acc1);
    }
    const float ivc = 0.07905694150f;   // 1/sqrt(160)
    int ont = (w < 4) ? nt : nt + 4;
    store_acc(c12oL,          128, lane, ont, acc0, ivc);
    store_acc(c12oL + 16*128, 128, lane, ont, acc1, ivc);
  }
  __syncthreads();

  // ---- Phase 3: silu/gates, rebuild gated v_out -> f16 ----
  for (int idx = tid; idx < 32*128; idx += 256) {
    int e = idx >> 7, d = idx & 127;
    float x = sOutL[e*256 + d];
    sActH[e*136 + d] = (_Float16)(x * sigm(x));
  }
  for (int idx = tid; idx < 32*64; idx += 256) {
    int e = idx >> 6, u = idx & 63;
    float gate = sigm(sOutL[e*256 + 128 + u]);
    float c0o  = sOutL[e*256 + 192 + u];
    float c1o  = c12oL[e*128 + u];
    float c2o  = c12oL[e*128 + 64 + u];
    const float* fr = frameL + e*12;
#pragma unroll
    for (int i = 0; i < 3; ++i) {
      float vo = c0o*fr[i] + c1o*fr[3+i] + c2o*fr[6+i];
      vgH[(e*3 + i)*72 + u] = (_Float16)(vo * gate);
    }
  }
  __syncthreads();

  // ---- Phase 4: WMMA  s2 = silu(s)@Wp0 ; w = lat@W_env ; v2 = vg@Wpv ----
  const float is128 = 0.08838834765f;  // 1/sqrt(128)
  {
    v8f acc0 = ZERO8, acc1 = ZERO8;
#pragma unroll
    for (int kt = 0; kt < 4; ++kt) {
      v16h b = load_b_frag(Wp0h, kt*8 + w, lane);
      acc0 = WMMA(load_a_frag(sActH,          136, lane, kt), b, acc0);
      acc1 = WMMA(load_a_frag(sActH + 16*136, 136, lane, kt), b, acc1);
    }
    store_acc(s2L,          128, lane, w, acc0, is128);
    store_acc(s2L + 16*128, 128, lane, w, acc1, is128);
  }
  {
    v8f acc0 = ZERO8, acc1 = ZERO8;
#pragma unroll
    for (int kt = 0; kt < 4; ++kt) {
      v16h b = load_b_frag(Wenvh, kt*12 + w, lane);
      acc0 = WMMA(load_a_frag(latH,          136, lane, kt), b, acc0);
      acc1 = WMMA(load_a_frag(latH + 16*136, 136, lane, kt), b, acc1);
    }
    store_acc(wL,          192, lane, w, acc0, is128);
    store_acc(wL + 16*192, 192, lane, w, acc1, is128);
    if (w < 4) {
      v8f acc2 = ZERO8, acc3 = ZERO8;
#pragma unroll
      for (int kt = 0; kt < 4; ++kt) {
        v16h b = load_b_frag(Wenvh, kt*12 + 8 + w, lane);
        acc2 = WMMA(load_a_frag(latH,          136, lane, kt), b, acc2);
        acc3 = WMMA(load_a_frag(latH + 16*136, 136, lane, kt), b, acc3);
      }
      store_acc(wL,          192, lane, 8 + w, acc2, is128);
      store_acc(wL + 16*192, 192, lane, 8 + w, acc3, is128);
    }
  }
  {
    int nt = w & 3, mtb = w >> 2;     // wave's single N-tile, 3 M-tiles share B
    v8f acc0 = ZERO8, acc1 = ZERO8, acc2 = ZERO8;
#pragma unroll
    for (int kt = 0; kt < 2; ++kt) {
      v16h b = load_b_frag(Wpvh, kt*4 + nt, lane);
      acc0 = WMMA(load_a_frag(vgH + (mtb+0)*16*72, 72, lane, kt), b, acc0);
      acc1 = WMMA(load_a_frag(vgH + (mtb+2)*16*72, 72, lane, kt), b, acc1);
      acc2 = WMMA(load_a_frag(vgH + (mtb+4)*16*72, 72, lane, kt), b, acc2);
    }
    store_acc(v2L + (mtb+0)*16*64, 64, lane, nt, acc0, 0.125f);
    store_acc(v2L + (mtb+2)*16*64, 64, lane, nt, acc1, 0.125f);
    store_acc(v2L + (mtb+4)*16*64, 64, lane, nt, acc2, 0.125f);
  }
  __syncthreads();

  // ---- Phase 5: env gating + scatter-add ----
  const float nrm = 0.25f;  // 1/sqrt(16)
  for (int idx = tid; idx < 32*128; idx += 256) {
    int e = idx >> 7, d = idx & 127;
    float val = (s2L[e*128 + d] + bp0[d]) * wL[e*192 + d] * nrm;
    atomicAdd(&sa[(size_t)dstL[e]*128 + d], val);
  }
  for (int idx = tid; idx < 32*192; idx += 256) {
    int e = idx / 192, q = idx % 192;
    int u = q / 3, i = q % 3;
    float val = v2L[(e*3 + i)*64 + u] * wL[e*192 + 128 + u] * nrm;
    atomicAdd(&va[(size_t)dstL[e]*192 + q], val);
  }
}

// ---- per-node output: class tensor products + output MLP + residual -------
__global__ __launch_bounds__(256) void node_out_kernel(
    const float* __restrict__ nf, const float* __restrict__ onehot,
    const float* __restrict__ sa, const float* __restrict__ va,
    const float* __restrict__ Wtp0, const float* __restrict__ Wtp1,
    const float* __restrict__ Wo0,  const float* __restrict__ bo0,
    const float* __restrict__ Wov,  const float* __restrict__ res_param,
    float* __restrict__ out) {
  __shared__ float saL[128], vaL[192], t0L[192], s3L[128], gL[64], v3L[192];
  __shared__ int clsS;
  int n = blockIdx.x, tid = threadIdx.x;
  if (tid == 0) {
    int c = 0; const float* oh = onehot + (size_t)n*16;
    for (int j = 0; j < 16; ++j) if (oh[j] > 0.5f) c = j;
    clsS = c;
  }
  for (int i = tid; i < 128; i += 256) saL[i] = sa[(size_t)n*128 + i];
  for (int i = tid; i < 192; i += 256) vaL[i] = va[(size_t)n*192 + i];
  __syncthreads();
  int c = clsS;
  const float* W0 = Wtp0 + (size_t)c*192;
  for (int v = tid; v < 192; v += 256) {
    float acc = 0.f;
    for (int u = 0; u < 128; ++u) acc += saL[u] * W0[(size_t)u*3072 + v];
    t0L[v] = acc * 0.02209708691f;  // 1/sqrt(128*16)
  }
  __syncthreads();
  for (int d = tid; d < 128; d += 256) { float x = t0L[d]; s3L[d] = x * sigm(x); }
  for (int u = tid; u < 64;  u += 256) gL[u] = sigm(t0L[128 + u]);
  __syncthreads();
  const float* W1 = Wtp1 + (size_t)c*64;
  for (int q = tid; q < 192; q += 256) {
    int v = q / 3, i = q % 3;
    float acc = 0.f;
    for (int u = 0; u < 64; ++u) acc += vaL[u*3 + i] * W1[(size_t)u*1024 + v];
    v3L[q] = acc * 0.03125f * gL[v];       // 1/sqrt(64*16)
  }
  __syncthreads();
  float r = sigm(res_param[0]);
  for (int d = tid; d < 128; d += 256) {
    float acc = 0.f;
    for (int u = 0; u < 128; ++u) acc += s3L[u] * Wo0[(size_t)u*128 + d];
    float s4 = acc * 0.08838834765f + bo0[d];
    out[(size_t)n*320 + d] = r * nf[(size_t)n*320 + d] + (1.f - r) * s4;
  }
  for (int q = tid; q < 192; q += 256) {
    int v = q / 3, i = q % 3;
    float acc = 0.f;
    for (int u = 0; u < 64; ++u) acc += v3L[u*3 + i] * Wov[(size_t)u*64 + v];
    float v4 = acc * 0.125f;
    out[(size_t)n*320 + 128 + q] = r * nf[(size_t)n*320 + 128 + q] + (1.f - r) * v4;
  }
}

// ---------------------------------------------------------------------------
extern "C" void kernel_launch(void* const* d_in, const int* in_sizes, int n_in,
                              void* d_out, int out_size, void* d_ws, size_t ws_size,
                              hipStream_t stream) {
  const float* node_feat = (const float*)d_in[0];
  const float* edge_feat = (const float*)d_in[1];
  const float* edge_vec  = (const float*)d_in[2];
  const float* latents   = (const float*)d_in[3];
  const float* onehot    = (const float*)d_in[4];
  const float* ln_n_g0   = (const float*)d_in[5];
  const float* ln_n_b0   = (const float*)d_in[6];
  const float* ln_n_g1   = (const float*)d_in[7];
  const float* ln_e_g0   = (const float*)d_in[8];
  const float* ln_e_b0   = (const float*)d_in[9];
  const float* ln_e_g1   = (const float*)d_in[10];
  const float* W_m0      = (const float*)d_in[11];
  const float* w_r       = (const float*)d_in[12];
  const float* w_i       = (const float*)d_in[13];
  const float* W_env     = (const float*)d_in[14];
  const float* Wp0       = (const float*)d_in[15];
  const float* bp0       = (const float*)d_in[16];
  const float* Wpv       = (const float*)d_in[17];
  const float* Wtp0      = (const float*)d_in[18];
  const float* Wtp1      = (const float*)d_in[19];
  const float* Wo0       = (const float*)d_in[20];
  const float* bo0       = (const float*)d_in[21];
  const float* Wov       = (const float*)d_in[22];
  const float* res_param = (const float*)d_in[23];
  const int*   edge_index= (const int*)  d_in[24];
  float* out = (float*)d_out;

  char* ws = (char*)d_ws;
  size_t off = 0;
  auto carve = [&](size_t bytes) { void* p = ws + off; off += (bytes + 255) & ~(size_t)255; return p; };
  float*    nodeS = (float*)carve((size_t)NN * 128 * 4);
  float*    nodeV = (float*)carve((size_t)NN * 192 * 4);
  float*    sa    = (float*)carve((size_t)NN * 128 * 4);
  float*    va    = (float*)carve((size_t)NN * 192 * 4);
  _Float16* Wm0h  = (_Float16*)carve((size_t)15*16*512 * 2);
  _Float16* wrh   = (_Float16*)carve((size_t)5*4*512 * 2);
  _Float16* wih   = (_Float16*)carve((size_t)5*4*512 * 2);
  _Float16* wihn  = (_Float16*)carve((size_t)5*4*512 * 2);
  _Float16* Wp0h  = (_Float16*)carve((size_t)4*8*512 * 2);
  _Float16* Wenvh = (_Float16*)carve((size_t)4*12*512 * 2);
  _Float16* Wpvh  = (_Float16*)carve((size_t)2*4*512 * 2);

  (void)hipMemsetAsync(sa, 0, (size_t)NN * 128 * 4, stream);
  (void)hipMemsetAsync(va, 0, (size_t)NN * 192 * 4, stream);

  auto cgrid = [](int total) { return dim3((total + 255) / 256); };
  convert_frag_kernel<<<cgrid(15*16*512), 256, 0, stream>>>(W_m0,  Wm0h, 15, 16, 256, 0);
  convert_frag_kernel<<<cgrid(5*4*512),   256, 0, stream>>>(w_r,   wrh,   5,  4,  64, 0);
  convert_frag_kernel<<<cgrid(5*4*512),   256, 0, stream>>>(w_i,   wih,   5,  4,  64, 0);
  convert_frag_kernel<<<cgrid(5*4*512),   256, 0, stream>>>(w_i,   wihn,  5,  4,  64, 1);
  convert_frag_kernel<<<cgrid(4*8*512),   256, 0, stream>>>(Wp0,   Wp0h,  4,  8, 128, 0);
  convert_frag_kernel<<<cgrid(4*12*512),  256, 0, stream>>>(W_env, Wenvh, 4, 12, 192, 0);
  convert_frag_kernel<<<cgrid(2*4*512),   256, 0, stream>>>(Wpv,   Wpvh,  2,  4,  64, 0);

  node_ln_kernel<<<NN/8, 256, 0, stream>>>(node_feat, ln_n_g0, ln_n_b0, ln_n_g1, nodeS, nodeV);

  edge_kernel<<<NE/32, 256, 0, stream>>>(edge_feat, edge_vec, latents,
                                         ln_e_g0, ln_e_b0, ln_e_g1, edge_index,
                                         nodeS, nodeV,
                                         Wm0h, wrh, wih, wihn, Wp0h, Wenvh, Wpvh, bp0,
                                         sa, va);

  node_out_kernel<<<NN, 256, 0, stream>>>(node_feat, onehot, sa, va,
                                          Wtp0, Wtp1, Wo0, bo0, Wov, res_param, out);
}
